// BitLinear_33449205301739
// MI455X (gfx1250) — 8-bit matrix operand order repaired and checked by reading, compiled, not run on hardware
//
#include <hip/hip_runtime.h>

typedef int v8i __attribute__((ext_vector_type(8)));

#define QEPS 1e-5f

__device__ __forceinline__ float waveReduceSum(float v) {
#pragma unroll
  for (int m = 16; m >= 1; m >>= 1) v += __shfl_xor(v, m, 32);
  return v;
}
__device__ __forceinline__ float waveReduceMax(float v) {
#pragma unroll
  for (int m = 16; m >= 1; m >>= 1) v = fmaxf(v, __shfl_xor(v, m, 32));
  return v;
}

__global__ void bl_abs_pass1(const float4* __restrict__ w4,
                             float* __restrict__ partials, int per_block) {
  int base = blockIdx.x * per_block;
  float s = 0.f;
  for (int j = threadIdx.x; j < per_block; j += 256) {
    float4 v = w4[base + j];
    s += fabsf(v.x) + fabsf(v.y) + fabsf(v.z) + fabsf(v.w);
  }
  s = waveReduceSum(s);
  __shared__ float sm[8];
  int lane = threadIdx.x & 31, wid = threadIdx.x >> 5;
  if (lane == 0) sm[wid] = s;
  __syncthreads();
  if (threadIdx.x == 0) {
    float t = 0.f;
#pragma unroll
    for (int i = 0; i < 8; ++i) t += sm[i];
    partials[blockIdx.x] = t;
  }
}

__global__ void bl_abs_pass2(const float* __restrict__ partials,
                             float* __restrict__ alpha_out, float count) {
  int tid = threadIdx.x;
  float s = partials[tid] + partials[tid + 256] + partials[tid + 512] +
            partials[tid + 768];
  s = waveReduceSum(s);
  __shared__ float sm[8];
  int lane = tid & 31, wid = tid >> 5;
  if (lane == 0) sm[wid] = s;
  __syncthreads();
  if (tid == 0) {
    float t = 0.f;
#pragma unroll
    for (int i = 0; i < 8; ++i) t += sm[i];
    alpha_out[0] = fmaxf(t / count, QEPS);
  }
}

__global__ void bl_quant_w(const float4* __restrict__ w4,
                           const float* __restrict__ alphap,
                           int* __restrict__ qw32) {
  int i = blockIdx.x * 256 + threadIdx.x;
  float a = alphap[0];
  float4 v = w4[i];
  int q0 = (int)fminf(fmaxf(rintf(v.x / a), -1.f), 1.f);
  int q1 = (int)fminf(fmaxf(rintf(v.y / a), -1.f), 1.f);
  int q2 = (int)fminf(fmaxf(rintf(v.z / a), -1.f), 1.f);
  int q3 = (int)fminf(fmaxf(rintf(v.w / a), -1.f), 1.f);
  qw32[i] = (q0 & 0xff) | ((q1 & 0xff) << 8) | ((q2 & 0xff) << 16) |
            ((q3 & 0xff) << 24);
}

__global__ void bl_quant_x(const float4* __restrict__ x4,
                           signed char* __restrict__ qx8,
                           float* __restrict__ scales, int K) {
  int token = blockIdx.x;
  int tid = threadIdx.x;
  const float4* row = x4 + (size_t)token * (K >> 2);
  float4 v0 = row[tid * 2 + 0];
  float4 v1 = row[tid * 2 + 1];
  float m = fmaxf(fmaxf(fmaxf(fabsf(v0.x), fabsf(v0.y)),
                        fmaxf(fabsf(v0.z), fabsf(v0.w))),
                  fmaxf(fmaxf(fabsf(v1.x), fabsf(v1.y)),
                        fmaxf(fabsf(v1.z), fabsf(v1.w))));
  m = waveReduceMax(m);
  __shared__ float sm[8];
  __shared__ float sscale;
  int lane = tid & 31, wid = tid >> 5;
  if (lane == 0) sm[wid] = m;
  __syncthreads();
  if (tid == 0) {
    float t = sm[0];
#pragma unroll
    for (int i = 1; i < 8; ++i) t = fmaxf(t, sm[i]);
    float sc = fmaxf(t, QEPS) * (1.0f / 127.0f);
    sscale = sc;
    scales[token] = sc;
  }
  __syncthreads();
  float s = sscale;
  float xs[8] = {v0.x, v0.y, v0.z, v0.w, v1.x, v1.y, v1.z, v1.w};
  int q[8];
#pragma unroll
  for (int j = 0; j < 8; ++j)
    q[j] = (int)fminf(fmaxf(rintf(xs[j] / s), -128.f), 127.f);
  int2 p;
  p.x = (q[0] & 0xff) | ((q[1] & 0xff) << 8) | ((q[2] & 0xff) << 16) |
        ((q[3] & 0xff) << 24);
  p.y = (q[4] & 0xff) | ((q[5] & 0xff) << 8) | ((q[6] & 0xff) << 16) |
        ((q[7] & 0xff) << 24);
  *(int2*)(qx8 + (size_t)token * K + tid * 8) = p;
}

#define A_STR 80
#define B_STR 80
#define A_BUF (128 * A_STR)
#define B_BUF (64 * B_STR)

__global__ __launch_bounds__(256) void bl_gemm(
    const signed char* __restrict__ qx8, const signed char* __restrict__ qw8,
    const float* __restrict__ scales, const float* __restrict__ alphap,
    const float* __restrict__ bias, float* __restrict__ out, int N, int K) {
  __shared__ signed char lA[2 * A_BUF];
  __shared__ signed char lB[2 * B_BUF];

  const int tid = threadIdx.x;
  const int lane = tid & 31;
  const int wave = tid >> 5;
  const int m0 = blockIdx.y * 128;
  const int n0 = blockIdx.x * 64;

  const int lr = tid >> 2;
  const int lc = (tid & 3) * 16;

  const int aRow = wave * 16 + (lane & 15);
  const int aKo = (lane >> 4) * 16;
  const int bRowBase = lane & 15;
  const int bKo = (lane >> 4) * 16;

  const signed char* Ag = qx8 + (size_t)m0 * K;
  const signed char* Bg = qw8 + (size_t)n0 * K;

  v8i acc[4] = {};
  const int NK = K >> 6;

  int4 bv = *(const int4*)(Bg + (size_t)lr * K + lc);
  int4 av0 = *(const int4*)(Ag + (size_t)lr * K + lc);
  int4 av1 = *(const int4*)(Ag + (size_t)(lr + 64) * K + lc);

  for (int kt = 0; kt < NK; ++kt) {
    const int bufA = (kt & 1) * A_BUF;
    const int bufB = (kt & 1) * B_BUF;
    *(int4*)(lB + bufB + lr * B_STR + lc) = bv;
    *(int4*)(lA + bufA + lr * A_STR + lc) = av0;
    *(int4*)(lA + bufA + (lr + 64) * A_STR + lc) = av1;
    __syncthreads();

    if (kt + 1 < NK) {
      const int k = (kt + 1) << 6;
      bv = *(const int4*)(Bg + (size_t)lr * K + k + lc);
      av0 = *(const int4*)(Ag + (size_t)lr * K + k + lc);
      av1 = *(const int4*)(Ag + (size_t)(lr + 64) * K + k + lc);
      if (k + 64 < K) {
        __builtin_prefetch(Bg + (size_t)lr * K + k + 64 + lc, 0, 3);
        __builtin_prefetch(Ag + (size_t)lr * K + k + 64 + lc, 0, 3);
        __builtin_prefetch(Ag + (size_t)(lr + 64) * K + k + 64 + lc, 0, 3);
      }
    }

    v8i a;
    const signed char* ap = lA + bufA + aRow * A_STR + aKo;
    ((int4*)&a)[0] = *(const int4*)(ap + 0);
    ((int4*)&a)[1] = *(const int4*)(ap + 32);

    v8i b[4];
#pragma unroll
    for (int t = 0; t < 4; ++t) {
      const signed char* bp = lB + bufB + (t * 16 + bRowBase) * B_STR + bKo;
      ((int4*)&b[t])[0] = *(const int4*)(bp + 0);
      ((int4*)&b[t])[1] = *(const int4*)(bp + 32);
    }
#pragma unroll
    for (int t = 0; t < 4; ++t)
      acc[t] = __builtin_amdgcn_wmma_i32_16x16x64_iu8(
           true, a,  true, b[t], acc[t],
           false,  false);
  }

  const float alpha = alphap[0];
  const int mbase = m0 + wave * 16 + (lane >> 4) * 8;
  float srow[8];
#pragma unroll
  for (int r = 0; r < 8; ++r) srow[r] = scales[mbase + r] * alpha;
#pragma unroll
  for (int t = 0; t < 4; ++t) {
    const int n = n0 + t * 16 + (lane & 15);
    const float bn = bias[n];
#pragma unroll
    for (int r = 0; r < 8; ++r)
      out[(size_t)(mbase + r) * N + n] = (float)acc[t][r] * srow[r] + bn;
  }
}

extern "C" void kernel_launch(void* const* d_in, const int* in_sizes, int n_in,
                              void* d_out, int out_size, void* d_ws,
                              size_t ws_size, hipStream_t stream) {
  const float* x = (const float*)d_in[0];
  const float* w = (const float*)d_in[1];
  const float* bias = (const float*)d_in[2];
  float* out = (float*)d_out;

  const int w_elems = in_sizes[1];
  const int D_OUT = in_sizes[2];
  const int D_IN = w_elems / D_OUT;
  const int M = in_sizes[0] / D_IN;

  char* ws = (char*)d_ws;
  float* wsAlpha = (float*)ws;
  float* partials = (float*)(ws + 256);
  float* scales = (float*)(ws + 8192);
  size_t off = 8192 + (size_t)M * 4;
  off = (off + 255) & ~(size_t)255;
  signed char* qw8 = (signed char*)(ws + off);
  off += (size_t)w_elems;
  off = (off + 255) & ~(size_t)255;
  signed char* qx8 = (signed char*)(ws + off);

  const int w4 = w_elems / 4;
  const int per_block = w4 / 1024;

  bl_abs_pass1<<<1024, 256, 0, stream>>>((const float4*)w, partials, per_block);
  bl_abs_pass2<<<1, 256, 0, stream>>>(partials, wsAlpha, (float)w_elems);
  bl_quant_w<<<w4 / 256, 256, 0, stream>>>((const float4*)w, wsAlpha,
                                           (int*)qw8);
  bl_quant_x<<<M, 256, 0, stream>>>((const float4*)x, qx8, scales, D_IN);

  dim3 grid(D_OUT / 64, M / 128);
  bl_gemm<<<grid, 256, 0, stream>>>(qx8, qw8, scales, wsAlpha, bias, out,
                                    D_OUT, D_IN);
}
